// ShiftedCausalSelfAttention_14637248545483
// MI455X (gfx1250) — compile-verified, matrix operand order checked by reading, not run on hardware
//
#include <hip/hip_runtime.h>

// ---------------------------------------------------------------------------
// ShiftedCausalSelfAttention for MI455X (gfx1250, wave32, WMMA 16x16x32 f16)
//
// B=2, T=1024, C=768, H=12, D=64.
// Memory floor ~230MB (prev_attn in + final_att out) => ~10us @ 23.3 TB/s.
// 16.4 GFLOP of GEMM work runs in f16 WMMA w/ f32 accumulate so compute stays
// below the bandwidth floor; softmax / shift-interp / blend stay in fp32.
// ---------------------------------------------------------------------------

#define BB 2
#define TT 1024
#define CC 768
#define HH 12
#define DD 64
#define NTOK (BB*TT)          // 2048
#define N3C  (3*CC)           // 2304

typedef __attribute__((ext_vector_type(16))) _Float16 v16h;
typedef __attribute__((ext_vector_type(8)))  _Float16 v8h;
typedef __attribute__((ext_vector_type(8)))  float    v8f;

__device__ __forceinline__ v8f wmma_f16(v16h a, v16h b, v8f c) {
  return __builtin_amdgcn_wmma_f32_16x16x32_f16(false, a, false, b, (short)0, c,
                                                false, false);
}

// Load a 16x32 f16 A/B fragment from a row-major array (rows 16B-aligned).
// ISA 7.12.2 16-bit A layout: lane L: m = L%16, half = L>>4;
// element e<8 -> K = half*8+e ; e>=8 -> K = 16+half*8+(e-8).
// `base` must already be offset by k0.
__device__ __forceinline__ v16h load_frag_rm(const _Float16* __restrict__ base,
                                             int ld, int row0) {
  int lane = threadIdx.x & 31;
  int r = lane & 15, half = lane >> 4;
  const _Float16* p = base + (size_t)(row0 + r) * ld + half * 8;
  v8h lo = *(const v8h*)(p);
  v8h hi = *(const v8h*)(p + 16);
  return __builtin_shufflevector(lo, hi, 0,1,2,3,4,5,6,7,8,9,10,11,12,13,14,15);
}

// Same fragment but sourced from fp32 LDS rows (ld = TT), converted to f16.
__device__ __forceinline__ v16h load_frag_lds_f32(const float* s, int k0) {
  int lane = threadIdx.x & 31;
  int m = lane & 15, half = lane >> 4;
  const float* p = s + m * TT + k0 + half * 8;
  v16h o;
  #pragma unroll
  for (int i = 0; i < 8; ++i) {
    o[i]     = (_Float16)p[i];
    o[8 + i] = (_Float16)p[16 + i];
  }
  return o;
}

// ------------------------- conversion kernels ------------------------------
__global__ void k_cvt_f16(const float* __restrict__ in, _Float16* __restrict__ out, int n) {
  int i = blockIdx.x * blockDim.x + threadIdx.x;
  if (i < n) out[i] = (_Float16)in[i];
}

// WT[n*K + k] = W[k*N + n]  (f16)
__global__ void k_transpose_cvt(const float* __restrict__ W, _Float16* __restrict__ WT,
                                int K, int N) {
  int o = blockIdx.x * blockDim.x + threadIdx.x;
  if (o >= K * N) return;
  int n = o / K, k = o % K;
  WT[o] = (_Float16)W[(size_t)k * N + n];
}

// ------------------------- K1: qkv GEMM ------------------------------------
// xh[2048x768] @ WqkvT(row n = col of Wqkv)[2304x768] + bqkv
// -> qh,kh [B,H,T,D] f16 ; v stored transposed vT [B,H,D,T] f16.
__global__ void __launch_bounds__(256)
k_gemm_qkv(const _Float16* __restrict__ xh, const _Float16* __restrict__ WT,
           const float* __restrict__ bias,
           _Float16* __restrict__ qh, _Float16* __restrict__ kh,
           _Float16* __restrict__ vT) {
  int gid  = blockIdx.x * 8 + (threadIdx.x >> 5);   // 4608 waves = 128 mt * 36 nt64
  int nt64 = gid % 36;
  int mt   = gid / 36;
  v8f acc[4] = {};
  for (int kk = 0; kk < CC; kk += 32) {
    v16h a = load_frag_rm(xh + kk, CC, mt * 16);
    #pragma unroll
    for (int t = 0; t < 4; ++t) {
      v16h b = load_frag_rm(WT + kk, CC, nt64 * 64 + t * 16);
      acc[t] = wmma_f16(a, b, acc[t]);
    }
  }
  int lane = threadIdx.x & 31, n = lane & 15, half = lane >> 4;
  #pragma unroll
  for (int t = 0; t < 4; ++t) {
    int col = nt64 * 64 + t * 16 + n;       // 0..2303
    float bv = bias[col];
    int which = col / CC;                   // 0=q 1=k 2=v
    int cc = col % CC;
    int h = cc >> 6, d = cc & 63;
    #pragma unroll
    for (int r = 0; r < 8; ++r) {
      int rowg = mt * 16 + r + 8 * half;    // token index 0..2047
      int b = rowg >> 10, tt = rowg & 1023;
      _Float16 hv = (_Float16)(acc[t][r] + bv);
      size_t bh = (size_t)(b * HH + h);
      if (which == 0)      qh[(bh * TT + tt) * DD + d] = hv;
      else if (which == 1) kh[(bh * TT + tt) * DD + d] = hv;
      else                 vT[(bh * DD + d) * TT + tt] = hv;
    }
  }
}

// ------------------------- K2: beta / gamma --------------------------------
// beta[b,h,t,i] = softmax_i( (x@Wattn+battn)[b,t,i*H+h] ), gamma = sigmoid(x@Wlayer+b)
__global__ void k_beta_gamma(const float* __restrict__ x,
                             const float* __restrict__ Wattn, const float* __restrict__ battn,
                             const float* __restrict__ Wlayer, const float* __restrict__ blayer,
                             float* __restrict__ beta, float* __restrict__ gamma) {
  int bt = blockIdx.x;                 // 0..2047
  int b = bt >> 10, t = bt & 1023;
  const float* xr = x + (size_t)bt * CC;
  __shared__ float l[7 * HH];
  int c = threadIdx.x;                 // 96 threads
  if (c < 7 * HH) {
    float a = battn[c];
    for (int k = 0; k < CC; ++k) a += xr[k] * Wattn[(size_t)k * (7 * HH) + c];
    l[c] = a;
  } else {
    int h = c - 7 * HH;                // 0..11
    float a = blayer[h];
    for (int k = 0; k < CC; ++k) a += xr[k] * Wlayer[(size_t)k * HH + h];
    gamma[((b * HH + h) << 10) + t] = 1.0f / (1.0f + __expf(-a));
  }
  __syncthreads();
  if (c < HH) {
    int h = c;
    float mx = -3.0e38f;
    #pragma unroll
    for (int i = 0; i < 7; ++i) mx = fmaxf(mx, l[i * HH + h]);
    float e[7], sum = 0.f;
    #pragma unroll
    for (int i = 0; i < 7; ++i) { e[i] = __expf(l[i * HH + h] - mx); sum += e[i]; }
    float inv = 1.0f / sum;
    float* bp = beta + ((size_t)(b * HH + h) * TT + t) * 7;
    #pragma unroll
    for (int i = 0; i < 7; ++i) bp[i] = e[i] * inv;
  }
}

// ------------------------- K3: fused attention -----------------------------
// Per block: (b,h) and 16 query rows. LDS holds 16x1024 fp32 score/att rows.
__global__ void __launch_bounds__(256)
k_attention(const _Float16* __restrict__ qh, const _Float16* __restrict__ kh,
            const _Float16* __restrict__ vT,
            const float* __restrict__ beta, const float* __restrict__ gamma,
            const float* __restrict__ prev,
            float* __restrict__ att_out, _Float16* __restrict__ yh) {
  __shared__ float s[16 * TT];                 // 64 KB
  int qt = blockIdx.x & 63;                    // 64 query tiles
  int bh = blockIdx.x >> 6;                    // 0..23
  int b = bh / HH, h = bh % HH;
  int tid = threadIdx.x, wave = tid >> 5;
  int qmax = qt * 16 + 15;

  const _Float16* qbase = qh + (size_t)bh * TT * DD;
  const _Float16* kbase = kh + (size_t)bh * TT * DD;
  const _Float16* vbase = vT + (size_t)bh * DD * TT;

  // --- phase 1: scores = (q k^T) / 8, WMMA, wave covers 128 key cols ---
  {
    v16h a0 = load_frag_rm(qbase +  0, DD, qt * 16);
    v16h a1 = load_frag_rm(qbase + 32, DD, qt * 16);
    int lane = tid & 31, n = lane & 15, half = lane >> 4;
    for (int nt = 0; nt < 8; ++nt) {
      int colbase = (wave * 8 + nt) * 16;
      if (colbase > qmax) break;               // fully masked tiles: skip (wave-uniform)
      v16h b0 = load_frag_rm(kbase +  0, DD, colbase);
      v16h b1 = load_frag_rm(kbase + 32, DD, colbase);
      v8f acc = {};
      acc = wmma_f16(a0, b0, acc);
      acc = wmma_f16(a1, b1, acc);
      #pragma unroll
      for (int r = 0; r < 8; ++r)
        s[(r + 8 * half) * TT + colbase + n] = acc[r] * 0.125f;
    }
  }
  __syncthreads();

  // --- phase 2: causal softmax per row (16 lanes per row, exact 0 above diag)
  int row = tid >> 4;                          // 0..15
  int lc  = tid & 15;
  int qr  = qt * 16 + row;
  float* srow = s + row * TT;
  {
    float mx = -3.0e38f;
    for (int c = lc; c <= qr; c += 16) mx = fmaxf(mx, srow[c]);
    #pragma unroll
    for (int off = 1; off < 16; off <<= 1) mx = fmaxf(mx, __shfl_xor(mx, off, 32));
    float sum = 0.f;
    for (int c = lc; c < TT; c += 16) {
      float e = 0.f;
      if (c <= qr) e = __expf(srow[c] - mx);
      srow[c] = e; sum += e;
    }
    #pragma unroll
    for (int off = 1; off < 16; off <<= 1) sum += __shfl_xor(sum, off, 32);
    float inv = 1.0f / sum;
    for (int c = lc; c < TT; c += 16) srow[c] *= inv;
  }
  __syncthreads();

  // --- phase 3: 7-shift interpolation + blend with prev_attn ---
  {
    float b7[7];
    const float* bp = beta + ((size_t)bh * TT + qr) * 7;
    #pragma unroll
    for (int i = 0; i < 7; ++i) b7[i] = bp[i];
    float g = gamma[bh * TT + qr];
    const float* prow = prev + ((size_t)(bh * TT + qr)) * TT;
    float* orow = att_out + ((size_t)(bh * TT + qr)) * TT;
    float fa[64];
    #pragma unroll
    for (int j = 0; j < 64; ++j) {
      int c = lc + 16 * j;
      float itp = b7[0] * srow[c];
      if (c + 1 < TT) itp += b7[1] * srow[c + 1];   // shift_left 1
      if (c - 1 >= 0) itp += b7[2] * srow[c - 1];   // shift_right 1
      if (c + 2 < TT) itp += b7[3] * srow[c + 2];
      if (c - 2 >= 0) itp += b7[4] * srow[c - 2];
      if (c + 3 < TT) itp += b7[5] * srow[c + 3];
      if (c - 3 >= 0) itp += b7[6] * srow[c - 3];
      if (c > qr) itp = 0.f;                        // causal zero on interp only
      float f = g * itp + (1.f - g) * prow[c];      // final_att (nonzero above diag!)
      orow[c] = f;
      fa[j] = f;
    }
    __syncthreads();
    #pragma unroll
    for (int j = 0; j < 64; ++j) srow[lc + 16 * j] = fa[j];
  }
  __syncthreads();

  // --- phase 4: y_head = final_att(16x1024) @ v(1024x64), split-K over 2 waves
  {
    int dtile = wave & 3;       // 4 tiles of 16 d-cols
    int khalf = wave >> 2;      // K halves of 512
    v8f acc = {};
    for (int kk = khalf * 512; kk < khalf * 512 + 512; kk += 32) {
      v16h af = load_frag_lds_f32(s, kk);
      v16h bf = load_frag_rm(vbase + kk, TT, dtile * 16);  // vT rows = d columns
      acc = wmma_f16(af, bf, acc);
    }
    __syncthreads();            // all waves done reading s
    int lane = tid & 31;
    float* red = s;             // reuse LDS for split-K reduction
    if (khalf == 1) {
      #pragma unroll
      for (int r = 0; r < 8; ++r) red[(dtile * 8 + r) * 32 + lane] = acc[r];
    }
    __syncthreads();
    if (khalf == 0) {
      int n = lane & 15, half = lane >> 4;
      #pragma unroll
      for (int r = 0; r < 8; ++r) {
        float v = acc[r] + red[(dtile * 8 + r) * 32 + lane];
        int t = qt * 16 + r + 8 * half;
        int col = h * DD + dtile * 16 + n;
        yh[((size_t)(b * TT + t)) * CC + col] = (_Float16)v;
      }
    }
  }
}

// ------------------------- K4: output projection ---------------------------
__global__ void __launch_bounds__(256)
k_gemm_proj(const _Float16* __restrict__ yh, const _Float16* __restrict__ WT,
            const float* __restrict__ bias, float* __restrict__ out) {
  int gid  = blockIdx.x * 8 + (threadIdx.x >> 5);  // 1536 waves = 128 mt * 12 nt64
  int nt64 = gid % 12;
  int mt   = gid / 12;
  v8f acc[4] = {};
  for (int kk = 0; kk < CC; kk += 32) {
    v16h a = load_frag_rm(yh + kk, CC, mt * 16);
    #pragma unroll
    for (int t = 0; t < 4; ++t) {
      v16h b = load_frag_rm(WT + kk, CC, nt64 * 64 + t * 16);
      acc[t] = wmma_f16(a, b, acc[t]);
    }
  }
  int lane = threadIdx.x & 31, n = lane & 15, half = lane >> 4;
  #pragma unroll
  for (int t = 0; t < 4; ++t) {
    int col = nt64 * 64 + t * 16 + n;
    float bv = bias[col];
    #pragma unroll
    for (int r = 0; r < 8; ++r) {
      int rowg = mt * 16 + r + 8 * half;
      out[(size_t)rowg * CC + col] = acc[t][r] + bv;
    }
  }
}

// ---------------------------------------------------------------------------
extern "C" void kernel_launch(void* const* d_in, const int* in_sizes, int n_in,
                              void* d_out, int out_size, void* d_ws, size_t ws_size,
                              hipStream_t stream) {
  (void)in_sizes; (void)n_in; (void)out_size; (void)ws_size;
  const float* x      = (const float*)d_in[0];
  const float* prev   = (const float*)d_in[1];
  const float* Wqkv   = (const float*)d_in[2];
  const float* bqkv   = (const float*)d_in[3];
  const float* Wattn  = (const float*)d_in[4];
  const float* battn  = (const float*)d_in[5];
  const float* Wlayer = (const float*)d_in[6];
  const float* blayer = (const float*)d_in[7];
  const float* Wproj  = (const float*)d_in[8];
  const float* bproj  = (const float*)d_in[9];

  float* y_out   = (float*)d_out;                       // [B,T,C]
  float* att_out = y_out + (size_t)NTOK * CC;           // [B,H,T,T]

  // workspace carve-up (all chunks naturally 256B aligned)
  char* w = (char*)d_ws;
  _Float16* xh     = (_Float16*)w; w += (size_t)NTOK * CC * 2;        // 3.0 MB
  _Float16* WqkvT  = (_Float16*)w; w += (size_t)N3C * CC * 2;         // 3.4 MB
  _Float16* WprojT = (_Float16*)w; w += (size_t)CC * CC * 2;          // 1.1 MB
  _Float16* qh     = (_Float16*)w; w += (size_t)BB * HH * TT * DD * 2;
  _Float16* kh     = (_Float16*)w; w += (size_t)BB * HH * TT * DD * 2;
  _Float16* vT     = (_Float16*)w; w += (size_t)BB * HH * DD * TT * 2;
  _Float16* yh     = (_Float16*)w; w += (size_t)NTOK * CC * 2;
  float*    beta   = (float*)w;    w += (size_t)BB * HH * TT * 7 * 4;
  float*    gamma  = (float*)w;    w += (size_t)BB * HH * TT * 4;     // total ~21 MB

  // 0) f16 conversions / weight transposes
  {
    int n = NTOK * CC;
    k_cvt_f16<<<(n + 255) / 256, 256, 0, stream>>>(x, xh, n);
  }
  k_transpose_cvt<<<(CC * N3C + 255) / 256, 256, 0, stream>>>(Wqkv, WqkvT, CC, N3C);
  k_transpose_cvt<<<(CC * CC + 255) / 256, 256, 0, stream>>>(Wproj, WprojT, CC, CC);

  // 1) qkv GEMM (WMMA), scatter to q/k/vT head layouts
  k_gemm_qkv<<<(128 * 36) / 8, 256, 0, stream>>>(xh, WqkvT, bqkv, qh, kh, vT);

  // 2) shift weights + layer gate
  k_beta_gamma<<<NTOK, 96, 0, stream>>>(x, Wattn, battn, Wlayer, blayer, beta, gamma);

  // 3) fused attention: scores->softmax->shift-interp->blend->final_att-> @v
  k_attention<<<BB * HH * 64, 256, 0, stream>>>(qh, kh, vT, beta, gamma, prev,
                                                att_out, yh);

  // 4) output projection
  k_gemm_proj<<<(128 * 12) / 8, 256, 0, stream>>>(yh, WprojT, bproj, y_out);
}
